// BFP_R3DNet_30073361007204
// MI455X (gfx1250) — compile-verified, matrix operand order checked by reading, not run on hardware
//
#include <hip/hip_runtime.h>

// ---------------------------------------------------------------------------
// BFP-quantized R3D-18 forward for gfx1250 (MI455X).
// Implicit-GEMM conv3d on v_wmma_f32_16x16x32_f16:
//   - A (im2col gather): branchless batched loads, software-pipelined through
//     a ping-pong LDS tile (1 barrier / K-step)
//   - B (weights) packed per-32-K fragment-native -> direct global b128 loads
// ---------------------------------------------------------------------------

typedef __attribute__((ext_vector_type(16))) _Float16 v16h;
typedef __attribute__((ext_vector_type(8)))  _Float16 v8h;
typedef __attribute__((ext_vector_type(8)))  float    v8f;

#define TM 64
#define TN 64
#define TK 32
#define APAD 16   // A row stride 48 f16 = 96B: keeps 16B/32B alignment, skews banks

__device__ __forceinline__ float bfp_apply(float x, float step) {
    // mantisa_bit = 8 -> mmax = 128; clip(round(x/step), -128, 127) * step
    float r = __builtin_rintf(x / step);
    r = fminf(fmaxf(r, -128.0f), 127.0f);
    return r * step;
}

// ---------------- elementwise kernels ----------------

__global__ void quant_f32_to_f16(const float* __restrict__ in,
                                 const int* __restrict__ erow,
                                 _Float16* __restrict__ out,
                                 int total, int C, int DHW) {
    int i = blockIdx.x * blockDim.x + threadIdx.x;
    if (i >= total) return;
    int c = ((unsigned)i / (unsigned)DHW) % (unsigned)C;
    float step = exp2f((float)erow[c]) * (1.0f / 128.0f);
    out[i] = (_Float16)bfp_apply(in[i], step);
}

__global__ void requant_f16(const _Float16* __restrict__ in,
                            const int* __restrict__ erow,
                            _Float16* __restrict__ out,
                            int total, int C, int DHW) {
    int i = blockIdx.x * blockDim.x + threadIdx.x;
    if (i >= total) return;
    int c = ((unsigned)i / (unsigned)DHW) % (unsigned)C;
    float step = exp2f((float)erow[c]) * (1.0f / 128.0f);
    out[i] = (_Float16)bfp_apply((float)in[i], step);
}

__global__ void add_relu_f16(const _Float16* __restrict__ a,
                             const _Float16* __restrict__ b,
                             _Float16* __restrict__ o, int total) {
    int i = blockIdx.x * blockDim.x + threadIdx.x;
    if (i >= total) return;
    float v = (float)a[i] + (float)b[i];
    o[i] = (_Float16)fmaxf(v, 0.0f);
}

__global__ void avgpool_f16(const _Float16* __restrict__ in,
                            float* __restrict__ out, int NC, int DHW) {
    int i = blockIdx.x * blockDim.x + threadIdx.x;
    if (i >= NC) return;
    const _Float16* p = in + (long long)i * DHW;
    float s = 0.0f;
    for (int j = 0; j < DHW; ++j) s += (float)p[j];
    out[i] = s / (float)DHW;
}

// ---------------- per-conv prep kernels ----------------

// Pack f32 OIDHW weights into fragment-native f16 layout:
//   Bpk[(kc*Co + co)*32 + kk]  where k = kc*32 + kk, k = ((ci*KD+z)*KH+y)*KW+x
__global__ void pack_weights(const float* __restrict__ w,
                             _Float16* __restrict__ Bw,
                             int Co, int Ci, int KD, int KH, int KW,
                             int K, int Kpad) {
    int i = blockIdx.x * blockDim.x + threadIdx.x;
    int tot = (Kpad >> 5) * Co * 32;
    if (i >= tot) return;
    int kk = i & 31;
    int t  = i >> 5;
    int co = (int)((unsigned)t % (unsigned)Co);
    int kc = (int)((unsigned)t / (unsigned)Co);
    int k  = kc * 32 + kk;
    float v = 0.0f;
    if (k < K) {
        int kx = k % KW; int r = k / KW;
        int ky = r % KH; r /= KH;
        int kz = r % KD; int ci = r / KD;
        v = w[(((co * Ci + ci) * KD + kz) * KH + ky) * KW + kx];
    }
    Bw[i] = (_Float16)v;
}

// k-decode table: off = ci*inChan + kz*inPlane + ky*Wi + kx ; pack = kz|ky<<10|kx<<20
__global__ void build_kdec(int2* __restrict__ kd,
                           int KD, int KH, int KW,
                           int Wi, int inPlane, int inChan,
                           int K, int Kpad) {
    int k = blockIdx.x * blockDim.x + threadIdx.x;
    if (k >= Kpad) return;
    int off = 0, pack = 1023;               // pad: kz=1023 always fails bounds
    if (k < K) {
        int kx = k % KW; int r = k / KW;
        int ky = r % KH; r /= KH;
        int kz = r % KD; int ci = r / KD;
        off  = ci * inChan + kz * inPlane + ky * Wi + kx;
        pack = kz | (ky << 10) | (kx << 20);
    }
    kd[k] = make_int2(off, pack);
}

// ---------------- implicit-GEMM conv3d with WMMA ----------------
// mode bits: 1 = relu AFTER quant, 2 = relu BEFORE quant, 4 = BFP quant

__global__ __launch_bounds__(128)
void conv3d_bfp_wmma(const _Float16* __restrict__ X,   // [N,Ci,Di,Hi,Wi] f16
                     const _Float16* __restrict__ Bw,  // packed weights
                     const int2* __restrict__ Kt,      // k-decode table [Kpad]
                     const float* __restrict__ bias,   // [Co]
                     const int* __restrict__ erow,     // 512 shared exponents
                     _Float16* __restrict__ O,         // [N,Co,Do,Ho,Wo] f16
                     int Nb, int Ci, int Di, int Hi, int Wi,
                     int Co, int Do, int Ho, int Wo,
                     int sd, int sh, int sw,
                     int pd, int ph, int pw,
                     int K, int Kp, int mode)
{
    __shared__ _Float16 As[2][TM][TK + APAD];   // ping-pong A tile [m][k]

    const int t    = threadIdx.x;
    const int lane = t & 31;
    const int wave = t >> 5;

    const int M     = Nb * Do * Ho * Wo;
    const int mBase = blockIdx.x * TM;
    const int nBase = blockIdx.y * TN;

    // --- per-thread A gather coordinates (fixed output pixel per row) ---
    const int arow  = t >> 1;          // 0..63
    const int akoff = (t & 1) * 16;    // 0 or 16
    const int am = mBase + arow;
    const bool am_ok = (am < M);
    unsigned r = am_ok ? (unsigned)am : 0u;
    const int aow = (int)(r % (unsigned)Wo); r /= (unsigned)Wo;
    const int aoh = (int)(r % (unsigned)Ho); r /= (unsigned)Ho;
    const int aod = (int)(r % (unsigned)Do);
    const int an  = (int)(r / (unsigned)Do);
    const int id0 = aod * sd - pd;
    const int ih0 = aoh * sh - ph;
    const int iw0 = aow * sw - pw;
    const int inPlane = Hi * Wi;
    const int inChan  = Di * inPlane;
    // baseOff + Kt.off == ci*inChan + id*inPlane + ih*Wi + iw (>=0 when in bounds)
    const int baseOff = id0 * inPlane + ih0 * Wi + iw0;
    const _Float16* Xn = X + (long long)an * Ci * inChan;

    // Branchless batched im2col gather of 16 f16 at k-range [kk0+akoff, +16)
    auto gather = [&](int kk0, v8h& glo, v8h& ghi) {
        int4 qa[8];
        const int4* kt4 = (const int4*)(Kt + kk0 + akoff);  // 128B aligned
        #pragma unroll
        for (int j = 0; j < 8; ++j) qa[j] = kt4[j];          // one load clause
        int  offs[16];
        int  okm[16];
        #pragma unroll
        for (int j = 0; j < 8; ++j) {
            int pk0 = qa[j].y, pk1 = qa[j].w;
            int ok0 = am_ok &
                ((unsigned)(id0 + (pk0 & 1023))          < (unsigned)Di) &
                ((unsigned)(ih0 + ((pk0 >> 10) & 1023))  < (unsigned)Hi) &
                ((unsigned)(iw0 + (pk0 >> 20))           < (unsigned)Wi);
            int ok1 = am_ok &
                ((unsigned)(id0 + (pk1 & 1023))          < (unsigned)Di) &
                ((unsigned)(ih0 + ((pk1 >> 10) & 1023))  < (unsigned)Hi) &
                ((unsigned)(iw0 + (pk1 >> 20))           < (unsigned)Wi);
            okm[2 * j]     = ok0;
            okm[2 * j + 1] = ok1;
            offs[2 * j]     = ok0 ? (baseOff + qa[j].x) : 0;  // 0 is always safe
            offs[2 * j + 1] = ok1 ? (baseOff + qa[j].z) : 0;
        }
        _Float16 raw[16];
        #pragma unroll
        for (int i = 0; i < 16; ++i) raw[i] = Xn[offs[i]];   // independent loads
        #pragma unroll
        for (int i = 0; i < 8; ++i) {
            glo[i] = okm[i]     ? raw[i]     : (_Float16)0.0f;
            ghi[i] = okm[8 + i] ? raw[8 + i] : (_Float16)0.0f;
        }
    };

    // wave sub-tile
    const int wM = (wave & 1) * 32;
    const int wN = (wave >> 1) * 32;
    v8f acc[2][2] = {};
    const int ko = (lane >> 4) * 8;

    v8h cur_lo, cur_hi;
    gather(0, cur_lo, cur_hi);
    int p = 0;

    for (int k0 = 0; k0 < Kp; k0 += TK) {
        *(v8h*)&As[p][arow][akoff]     = cur_lo;
        *(v8h*)&As[p][arow][akoff + 8] = cur_hi;
        __syncthreads();

        // issue next K-step's gather loads early (overlap with WMMA below)
        v8h nlo = {}, nhi = {};
        if (k0 + TK < Kp) gather(k0 + TK, nlo, nhi);

        // ---- A fragments from LDS ----
        v16h af[2], bf[2];
        #pragma unroll
        for (int mt = 0; mt < 2; ++mt) {
            const int row = wM + mt * 16 + (lane & 15);
            v8h alo = *(const v8h*)&As[p][row][ko];
            v8h ahi = *(const v8h*)&As[p][row][16 + ko];
            af[mt] = __builtin_shufflevector(alo, ahi,
                     0,1,2,3,4,5,6,7,8,9,10,11,12,13,14,15);
        }
        // ---- B fragments direct from global (fragment-native packing) ----
        {
            const _Float16* bbase =
                Bw + (long long)(k0 >> 5) * Co * 32 + (long long)(nBase + wN) * 32;
            #pragma unroll
            for (int nt = 0; nt < 2; ++nt) {
                const _Float16* bp = bbase + (nt * 16 + (lane & 15)) * 32;
                v8h blo = *(const v8h*)(bp + ko);
                v8h bhi = *(const v8h*)(bp + 16 + ko);
                __builtin_prefetch(bp + (long long)Co * 32, 0, 1); // next K chunk
                bf[nt] = __builtin_shufflevector(blo, bhi,
                         0,1,2,3,4,5,6,7,8,9,10,11,12,13,14,15);
            }
        }
        #pragma unroll
        for (int mt = 0; mt < 2; ++mt)
            #pragma unroll
            for (int nt = 0; nt < 2; ++nt)
                acc[mt][nt] = __builtin_amdgcn_wmma_f32_16x16x32_f16(
                    false, af[mt], false, bf[nt],
                    (short)0, acc[mt][nt], false, false);

        cur_lo = nlo; cur_hi = nhi; p ^= 1;
    }

    // ---- fused epilogue: bias + relu/quant orders, division-free decode ----
    float bv[2], stepv[2]; int ncol[2];
    #pragma unroll
    for (int nt = 0; nt < 2; ++nt) {
        ncol[nt]  = nBase + wN + nt * 16 + (lane & 15);
        bv[nt]    = bias[ncol[nt]];
        stepv[nt] = (mode & 4) ? exp2f((float)erow[ncol[nt]]) * (1.0f / 128.0f)
                               : 1.0f;
    }
    #pragma unroll
    for (int mt = 0; mt < 2; ++mt) {
        const int base = mBase + wM + mt * 16 + ((lane >> 4) << 3);
        unsigned q = (unsigned)(base < M ? base : 0);
        int ow = (int)(q % (unsigned)Wo); q /= (unsigned)Wo;
        int oh = (int)(q % (unsigned)Ho); q /= (unsigned)Ho;
        int od = (int)(q % (unsigned)Do);
        int nb = (int)(q / (unsigned)Do);
        #pragma unroll
        for (int rr8 = 0; rr8 < 8; ++rr8) {
            if (base + rr8 < M) {
                const int pixBase = ((nb * Co) * Do + od) * Ho * Wo + oh * Wo + ow;
                #pragma unroll
                for (int nt = 0; nt < 2; ++nt) {
                    float v = acc[mt][nt][rr8] + bv[nt];
                    if (mode & 2) v = fmaxf(v, 0.0f);
                    if (mode & 4) v = bfp_apply(v, stepv[nt]);
                    if (mode & 1) v = fmaxf(v, 0.0f);
                    O[pixBase + ncol[nt] * (Do * Ho * Wo)] = (_Float16)v;
                }
            }
            // increment output pixel coords (wrap, no division)
            if (++ow == Wo) { ow = 0;
                if (++oh == Ho) { oh = 0;
                    if (++od == Do) { od = 0; ++nb; } } }
        }
    }
}

// ---------------- host orchestration ----------------

extern "C" void kernel_launch(void* const* d_in, const int* in_sizes, int n_in,
                              void* d_out, int out_size, void* d_ws, size_t ws_size,
                              hipStream_t stream) {
    (void)in_sizes; (void)n_in; (void)out_size; (void)ws_size;

    const float* x    = (const float*)d_in[0];
    const int*   exps = (const int*)d_in[1];

    // workspace carve (f16 activation buffers + packed weights + k-table)
    const size_t ACT_CAP = 26ull * 1024 * 1024;           // f16 elems (>= 25.69M)
    char* ws = (char*)d_ws;
    _Float16* bufX = (_Float16*)ws; ws += ACT_CAP * 2;
    _Float16* bufT = (_Float16*)ws; ws += ACT_CAP * 2;
    _Float16* bufR = (_Float16*)ws; ws += ACT_CAP * 2;
    _Float16* bufW = (_Float16*)ws; ws += 15ull * 1024 * 1024; // >=13824*512 f16
    int2*     bufD = (int2*)ws;                           // >= 13824 entries

    auto launch_conv = [&](const _Float16* in, _Float16* out,
                           const float* w, const float* b,
                           int eidx, int mode,
                           int Nb, int Ci, int Di, int Hi, int Wi, int Co,
                           int KD, int KH, int KW,
                           int sd, int sh, int sw, int pd, int ph, int pw,
                           int& Do, int& Ho, int& Wo) {
        Do = (Di + 2 * pd - KD) / sd + 1;
        Ho = (Hi + 2 * ph - KH) / sh + 1;
        Wo = (Wi + 2 * pw - KW) / sw + 1;
        int K  = Ci * KD * KH * KW;
        int Kp = (K + TK - 1) & ~(TK - 1);
        int inPlane = Hi * Wi, inChan = Di * inPlane;
        int ptot = Kp * Co;
        pack_weights<<<(ptot + 255) / 256, 256, 0, stream>>>(
            w, bufW, Co, Ci, KD, KH, KW, K, Kp);
        build_kdec<<<(Kp + 255) / 256, 256, 0, stream>>>(
            bufD, KD, KH, KW, Wi, inPlane, inChan, K, Kp);
        int M = Nb * Do * Ho * Wo;
        dim3 grid((unsigned)((M + TM - 1) / TM), (unsigned)(Co / TN));
        conv3d_bfp_wmma<<<grid, 128, 0, stream>>>(
            in, bufW, bufD, b, exps + (long long)eidx * 512, out,
            Nb, Ci, Di, Hi, Wi, Co, Do, Ho, Wo,
            sd, sh, sw, pd, ph, pw, K, Kp, mode);
    };

    // ---- stem: quant(exps0) -> conv1 -> relu -> quant(exps1) ----
    int N = 8, C = 3, D = 16, H = 112, W = 112;
    int tot = N * C * D * H * W;
    quant_f32_to_f16<<<(tot + 255) / 256, 256, 0, stream>>>(
        x, exps, bufT, tot, C, D * H * W);
    int Do, Ho, Wo;
    launch_conv(bufT, bufX, (const float*)d_in[2], (const float*)d_in[3],
                /*eidx=*/1, /*mode: relu-before + quant*/ 2 | 4,
                N, 3, D, H, W, 64, 3, 7, 7, 1, 2, 2, 1, 3, 3, Do, Ho, Wo);
    C = 64; D = Do; H = Ho; W = Wo;

    // ---- residual stages ----
    const int widths[4] = {64, 128, 256, 512};
    int pidx = 4, eidx = 2;
    for (int li = 0; li < 4; ++li) {
        int cout = widths[li];
        for (int bi = 0; bi < 2; ++bi) {
            bool ds = (li > 0 && bi == 0);
            int s = ds ? 2 : 1;
            const float* w1 = (const float*)d_in[pidx++];
            const float* b1 = (const float*)d_in[pidx++];
            const float* w2 = (const float*)d_in[pidx++];
            const float* b2 = (const float*)d_in[pidx++];
            const float* wd = nullptr; const float* bd = nullptr;
            if (ds) { wd = (const float*)d_in[pidx++];
                      bd = (const float*)d_in[pidx++]; }

            // res = relu(quant(conv1(x), exps[eidx]))
            launch_conv(bufX, bufT, w1, b1, eidx, 4 | 1,
                        N, C, D, H, W, cout, 3, 3, 3, s, s, s, 1, 1, 1,
                        Do, Ho, Wo);
            // res = quant(conv2(res), exps[eidx+1])
            int D2, H2, W2;
            launch_conv(bufT, bufR, w2, b2, eidx + 1, 4,
                        N, cout, Do, Ho, Wo, cout, 3, 3, 3, 1, 1, 1, 1, 1, 1,
                        D2, H2, W2);
            // x path: ds-conv + quant, or pure requant
            if (ds) {
                launch_conv(bufX, bufT, wd, bd, eidx + 1, 4,
                            N, C, D, H, W, cout, 1, 1, 1, 2, 2, 2, 0, 0, 0,
                            D2, H2, W2);
            } else {
                int t2 = N * C * D * H * W;
                requant_f16<<<(t2 + 255) / 256, 256, 0, stream>>>(
                    bufX, exps + (long long)(eidx + 1) * 512, bufT, t2, C,
                    D * H * W);
            }
            // x = relu(xq + res)
            int t3 = N * cout * D2 * H2 * W2;
            add_relu_f16<<<(t3 + 255) / 256, 256, 0, stream>>>(
                bufT, bufR, bufX, t3);
            C = cout; D = D2; H = H2; W = W2;
            eidx += 2;
        }
    }

    // ---- adaptive avg pool -> [8, 512] f32 ----
    avgpool_f16<<<(N * 512 + 255) / 256, 256, 0, stream>>>(
        bufX, (float*)d_out, N * 512, D * H * W);
}
